// AttLSTM_8899172237963
// MI455X (gfx1250) — matrix operand order repaired and checked by reading, not compiled, not run on hardware
//
#include <hip/hip_runtime.h>
#include <stdint.h>
#include <stddef.h>

// ---------------------------------------------------------------------------
// 3-layer LSTM (B=64,T=512,D=256,H=1024) + FC(1024->128)+ELU for MI455X.
// Persistent cooperative kernel, bf16 WMMA (v_wmma_f32_16x16x32_bf16),
// fp32 accumulation/state. Weights (44MB bf16) stay resident in 192MB L2.
// A-matrix slabs staged with GLOBAL_LOAD_ASYNC_TO_LDS_B128 (ASYNCcnt),
// double-buffered in LDS so staging overlaps WMMA compute.
// ---------------------------------------------------------------------------

#define B_   64
#define T_   512
#define D_   256
#define H_   1024
#define C_   128
#define N4_  4096          // 4*H
#define NWG  64
#define NTHR 256
#define BH   (B_ * H_)
#define KEEP 0.9f

typedef __attribute__((ext_vector_type(16))) __bf16 v16bf;
typedef __attribute__((ext_vector_type(8)))  __bf16 v8bf;
typedef __attribute__((ext_vector_type(8)))  float  v8f;

__device__ __forceinline__ unsigned short f2bf(float f) {
  unsigned u = __float_as_uint(f);
  return (unsigned short)((u + 0x7FFFu + ((u >> 16) & 1u)) >> 16);
}

__device__ __forceinline__ float sigm(float x) {
  return 1.0f / (1.0f + __expf(-x));
}

// ---------------- weight pack: fp32 row-major [K x 4H] -> bf16 col-major ----
__global__ void pack_w(const float* __restrict__ src,
                       unsigned short* __restrict__ dst, int K) {
  size_t total = (size_t)K * N4_;
  size_t stride = (size_t)gridDim.x * blockDim.x;
  for (size_t i = (size_t)blockIdx.x * blockDim.x + threadIdx.x; i < total; i += stride) {
    size_t n = i / (size_t)K;
    size_t k = i - n * (size_t)K;
    dst[i] = f2bf(src[k * N4_ + n]);   // dst[n*K + k] = bf16(W[k][n])
  }
}

__global__ void init_bar(unsigned* bar) {
  if (threadIdx.x < 2) bar[threadIdx.x] = 0u;
}

// ---------------- sense-reversing grid barrier ------------------------------
__device__ __forceinline__ void gbar(unsigned* bar) {
  __threadfence();
  __syncthreads();
  if (threadIdx.x == 0) {
    unsigned g = __hip_atomic_load(bar + 1, __ATOMIC_RELAXED, __HIP_MEMORY_SCOPE_AGENT);
    unsigned a = __hip_atomic_fetch_add(bar, 1u, __ATOMIC_ACQ_REL, __HIP_MEMORY_SCOPE_AGENT);
    if (a == NWG - 1u) {
      __hip_atomic_store(bar, 0u, __ATOMIC_RELAXED, __HIP_MEMORY_SCOPE_AGENT);
      __hip_atomic_fetch_add(bar + 1, 1u, __ATOMIC_RELEASE, __HIP_MEMORY_SCOPE_AGENT);
    } else {
      while (__hip_atomic_load(bar + 1, __ATOMIC_ACQUIRE, __HIP_MEMORY_SCOPE_AGENT) == g) {
        __builtin_amdgcn_s_sleep(2);
      }
    }
  }
  __syncthreads();
}

// ---- async global->LDS staging of one 64x128 bf16 A-slab (16KB) ------------
// Each thread moves 64B (4 x b128) of one row-run; per wave that is 4 async
// instructions tracked by ASYNCcnt.
__device__ __forceinline__ void stage_async(const unsigned short* src, int rs, int kb,
                                            unsigned short* lds_buf, int tid) {
  int e0 = tid * 32;                 // 32 consecutive bf16 elems, single row
  int bb = e0 >> 7;
  int kk = e0 & 127;
  unsigned long long g = (unsigned long long)(uintptr_t)(src + (size_t)bb * rs + kb + kk);
  unsigned l = (unsigned)(uintptr_t)(lds_buf + e0);   // low 32 bits = LDS offset
  asm volatile("global_load_async_to_lds_b128 %0, %1, off\n\t"
               "global_load_async_to_lds_b128 %0, %1, off offset:16\n\t"
               "global_load_async_to_lds_b128 %0, %1, off offset:32\n\t"
               "global_load_async_to_lds_b128 %0, %1, off offset:48"
               :: "v"(l), "v"(g) : "memory");
}

__device__ __forceinline__ void slab_src(int slab,
                                         const unsigned short* seg0, int rs0, int K0,
                                         const unsigned short* seg1, int rs1,
                                         const unsigned short*& s, int& rs, int& kb) {
  int kc = slab << 7;
  if (kc < K0) { s = seg0; rs = rs0; kb = kc; }
  else         { s = seg1; rs = rs1; kb = kc - K0; }
}

// ---------------- one LSTM cell: GEMM (WMMA) + gating -----------------------
// A = [seg0 | seg1] (B_ rows, Ktot cols, bf16), W packed col-major bf16.
// WG blockIdx.x owns hidden units [16w, 16w+16); wave v -> gate v>>1,
// M-blocks {2*(v&1), 2*(v&1)+1}.
__device__ __forceinline__ void cell_phase(
    const unsigned short* __restrict__ seg0, int rs0, int K0,
    const unsigned short* __restrict__ seg1, int rs1,
    int Ktot,
    const unsigned short* __restrict__ wpk,
    const float* __restrict__ bias,
    float* __restrict__ hst, float* __restrict__ cst,
    unsigned short* __restrict__ hbw,
    unsigned short* lds_a0, unsigned short* lds_a1, float* lds_pre) {

  const int tid   = threadIdx.x;
  const int lane  = tid & 31;
  const int wave  = tid >> 5;
  const int g_id  = wave >> 1;          // gate 0..3 = f,i,o,g
  const int m0    = (wave & 1) << 1;    // M-block pair
  const int nl    = lane & 15;
  const int khalf = lane >> 4;          // 0: lanes 0-15, 1: lanes 16-31
  const int wgu   = blockIdx.x * 16;    // first hidden unit of this WG

  const size_t ncol = (size_t)(g_id * H_ + wgu + nl);
  const unsigned short* wcol = wpk + ncol * (size_t)Ktot;

  v8f acc0 = {};
  v8f acc1 = {};

  const int nslab = Ktot >> 7;

  // prime the pipeline: slab 0 -> buffer 0
  {
    const unsigned short* s; int rs, kb;
    slab_src(0, seg0, rs0, K0, seg1, rs1, s, rs, kb);
    stage_async(s, rs, kb, lds_a0, tid);
  }

  #pragma unroll 1
  for (int i = 0; i < nslab; ++i) {
    unsigned short* cur = (i & 1) ? lds_a1 : lds_a0;
    unsigned short* nxt = (i & 1) ? lds_a0 : lds_a1;

    // all waves done computing on `nxt` (used 2 slabs ago) before overwriting
    __syncthreads();
    if (i + 1 < nslab) {
      const unsigned short* s; int rs, kb;
      slab_src(i + 1, seg0, rs0, K0, seg1, rs1, s, rs, kb);
      stage_async(s, rs, kb, nxt, tid);
      // keep next slab's weight columns warm while we compute
      __builtin_prefetch(wcol + ((i + 1) << 7) + khalf * 16, 0, 0);
      asm volatile("s_wait_asynccnt 0x4" ::: "memory");  // slab i complete
    } else {
      asm volatile("s_wait_asynccnt 0x0" ::: "memory");
    }
    __syncthreads();   // slab i visible to all waves

    const int kc = i << 7;
    #pragma unroll
    for (int ks = 0; ks < 128; ks += 32) {
      // B fragment: 32(K) x 16(N), per-lane 16 contiguous bf16 (32B) from L2
      v16bf Bf = *(const v16bf*)(wcol + (kc + ks) + khalf * 16);

      // A fragment for M-block m0: lane row = 16*m0 + nl, two 16B runs
      const unsigned short* a0p = cur + ((m0 * 16 + nl) * 128 + ks + khalf * 16);
      v8bf a0lo = *(const v8bf*)(a0p);
      v8bf a0hi = *(const v8bf*)(a0p + 8);
      v16bf A0 = __builtin_shufflevector(a0lo, a0hi,
                   0,1,2,3,4,5,6,7,8,9,10,11,12,13,14,15);
      const unsigned short* a1p = a0p + 16 * 128;
      v8bf a1lo = *(const v8bf*)(a1p);
      v8bf a1hi = *(const v8bf*)(a1p + 8);
      v16bf A1 = __builtin_shufflevector(a1lo, a1hi,
                   0,1,2,3,4,5,6,7,8,9,10,11,12,13,14,15);

      acc0 = __builtin_amdgcn_wmma_f32_16x16x32_bf16(
                 false, A0, false, Bf, (short)0, acc0, false, false);
      acc1 = __builtin_amdgcn_wmma_f32_16x16x32_bf16(
                 false, A1, false, Bf, (short)0, acc1, false, false);
    }
  }

  // ---- bias + spill pre-activations to LDS (64 batch x 64 gate-cols)
  const float bv = bias[g_id * H_ + wgu + nl];
  #pragma unroll
  for (int j = 0; j < 8; ++j) {
    int r = m0 * 16 + khalf * 8 + j;                 // batch row of acc0
    lds_pre[r * 64 + g_id * 16 + nl]        = acc0[j] + bv;
    lds_pre[(r + 16) * 64 + g_id * 16 + nl] = acc1[j] + bv;
  }
  __syncthreads();

  // ---- gating + recurrent-dropout mixing (eval form), state update
  for (int q = tid; q < B_ * 16; q += NTHR) {
    int bb = q >> 4;
    int ul = q & 15;
    float f  = lds_pre[bb * 64 +      ul];
    float ii = lds_pre[bb * 64 + 16 + ul];
    float oo = lds_pre[bb * 64 + 32 + ul];
    float gg = lds_pre[bb * 64 + 48 + ul];
    size_t idx = (size_t)bb * H_ + (wgu + ul);
    float co = cst[idx];
    float ho = hst[idx];
    float cn = sigm(f) * co + sigm(ii) * tanhf(gg);
    float hn = sigm(oo) * tanhf(cn);
    hn = hn * KEEP + ho * (1.0f - KEEP);
    cn = cn * KEEP + co * (1.0f - KEEP);
    hst[idx] = hn;
    cst[idx] = cn;
    hbw[idx] = f2bf(hn);
  }
}

// ---------------- persistent cooperative LSTM kernel ------------------------
__global__ __launch_bounds__(NTHR, 1)
void lstm_persist(const float* __restrict__ xin,
                  const unsigned short* __restrict__ wpk1, const float* __restrict__ b1,
                  const unsigned short* __restrict__ wpk2, const float* __restrict__ b2,
                  const unsigned short* __restrict__ wpk3, const float* __restrict__ b3,
                  const float* __restrict__ fcw, const float* __restrict__ fcb,
                  unsigned short* __restrict__ hb1,
                  unsigned short* __restrict__ hb2,
                  unsigned short* __restrict__ hb3,
                  float* __restrict__ h1, float* __restrict__ c1,
                  float* __restrict__ h2, float* __restrict__ c2,
                  float* __restrict__ h3, float* __restrict__ c3,
                  unsigned short* __restrict__ xb,
                  unsigned* bar, float* __restrict__ out) {

  __shared__ __align__(16) unsigned short lds_a0[64 * 128];  // 16 KB
  __shared__ __align__(16) unsigned short lds_a1[64 * 128];  // 16 KB
  __shared__ __align__(16) float lds_pre[64 * 64];           // 16 KB

  const int gtid = blockIdx.x * NTHR + threadIdx.x;

  // zero-init states (ws is poisoned; must be deterministic per call)
  for (int i = gtid; i < BH; i += NWG * NTHR) {
    h1[i] = 0.f; c1[i] = 0.f;
    h2[i] = 0.f; c2[i] = 0.f;
    h3[i] = 0.f; c3[i] = 0.f;
    hb1[i] = 0; hb1[i + BH] = 0;
    hb2[i] = 0; hb2[i + BH] = 0;
    hb3[i] = 0; hb3[i + BH] = 0;
  }
  gbar(bar);

  #pragma unroll 1
  for (int t = 0; t < T_; ++t) {
    const int rp = t & 1;        // read parity of old h
    const int wp = rp ^ 1;       // write parity of new h

    // convert x_t -> bf16 (one element per thread: grid size == B_*D_)
    {
      int i = gtid;
      int bb = i >> 8, dd = i & 255;
      xb[i] = f2bf(xin[(size_t)bb * T_ * D_ + (size_t)t * D_ + dd]);
    }
    gbar(bar);

    // cell 1: A = [h1_old | x_t], K = 1280
    cell_phase(hb1 + (size_t)rp * BH, H_, H_, xb, D_, H_ + D_,
               wpk1, b1, h1, c1, hb1 + (size_t)wp * BH, lds_a0, lds_a1, lds_pre);
    gbar(bar);

    // cell 2: A = [h2_old | h1_new], K = 2048
    cell_phase(hb2 + (size_t)rp * BH, H_, H_, hb1 + (size_t)wp * BH, H_, 2 * H_,
               wpk2, b2, h2, c2, hb2 + (size_t)wp * BH, lds_a0, lds_a1, lds_pre);
    gbar(bar);

    // cell 3: A = [h3_old | h2_new], K = 2048
    cell_phase(hb3 + (size_t)rp * BH, H_, H_, hb2 + (size_t)wp * BH, H_, 2 * H_,
               wpk3, b3, h3, c3, hb3 + (size_t)wp * BH, lds_a0, lds_a1, lds_pre);
  }
  gbar(bar);

  // final FC (64x128 <- 64x1024 @ 1024x128^T) + ELU, fp32
  for (int q = gtid; q < B_ * C_; q += NWG * NTHR) {
    int bb = q >> 7, cc = q & 127;
    const float4* hv = (const float4*)(h3 + (size_t)bb * H_);
    const float4* wv = (const float4*)(fcw + (size_t)cc * H_);
    float acc = fcb[cc];
    for (int k = 0; k < H_ / 4; ++k) {
      float4 a = hv[k], b = wv[k];
      acc += a.x * b.x + a.y * b.y + a.z * b.z + a.w * b.w;
    }
    out[q] = acc > 0.f ? acc : (__expf(acc) - 1.f);
  }
}

// ---------------------------------------------------------------------------
extern "C" void kernel_launch(void* const* d_in, const int* in_sizes, int n_in,
                              void* d_out, int out_size, void* d_ws, size_t ws_size,
                              hipStream_t stream) {
  const float* xin  = (const float*)d_in[0];
  const float* W1f  = (const float*)d_in[1];
  const float* b1f  = (const float*)d_in[2];
  const float* W2f  = (const float*)d_in[3];
  const float* b2f  = (const float*)d_in[4];
  const float* W3f  = (const float*)d_in[5];
  const float* b3f  = (const float*)d_in[6];
  const float* fcw  = (const float*)d_in[7];
  const float* fcb  = (const float*)d_in[8];
  float* out = (float*)d_out;

  char* ws = (char*)d_ws;
  size_t off = 0;
  auto carve = [&](size_t bytes) -> void* {
    void* p = ws + off;
    off += (bytes + 255) & ~(size_t)255;
    return p;
  };
  unsigned short* wpk1 = (unsigned short*)carve((size_t)N4_ * 1280 * 2);
  unsigned short* wpk2 = (unsigned short*)carve((size_t)N4_ * 2048 * 2);
  unsigned short* wpk3 = (unsigned short*)carve((size_t)N4_ * 2048 * 2);
  unsigned short* hb1  = (unsigned short*)carve((size_t)2 * BH * 2);
  unsigned short* hb2  = (unsigned short*)carve((size_t)2 * BH * 2);
  unsigned short* hb3  = (unsigned short*)carve((size_t)2 * BH * 2);
  float* h1 = (float*)carve((size_t)BH * 4);
  float* c1 = (float*)carve((size_t)BH * 4);
  float* h2 = (float*)carve((size_t)BH * 4);
  float* c2 = (float*)carve((size_t)BH * 4);
  float* h3 = (float*)carve((size_t)BH * 4);
  float* c3 = (float*)carve((size_t)BH * 4);
  unsigned short* xb = (unsigned short*)carve((size_t)B_ * D_ * 2);
  unsigned* bar = (unsigned*)carve(256);

  init_bar<<<1, 64, 0, stream>>>(bar);
  pack_w<<<2048, 256, 0, stream>>>(W1f, wpk1, 1280);
  pack_w<<<2048, 256, 0, stream>>>(W2f, wpk2, 2048);
  pack_w<<<2048, 256, 0, stream>>>(W3f, wpk3, 2048);

  lstm_persist<<<NWG, NTHR, 0, stream>>>(
      xin, wpk1, b1f, wpk2, b2f, wpk3, b3f, fcw, fcb,
      hb1, hb2, hb3, h1, c1, h2, c2, h3, c3, xb, bar, out);
}
